// XaiYoloWrapper_26139170964058
// MI455X (gfx1250) — compile-verified, matrix operand order checked by reading, not run on hardware
//
#include <hip/hip_runtime.h>
#include <hip/hip_bf16.h>
#include <float.h>

// Problem constants (match reference)
#define BATCH   32
#define NPRED   25200
#define NCLS    80
#define ROWSZ   85
#define CONF_T  0.25f
#define IOU_T   0.45f
#define MAXDET  1000
#define MAX_WH  7680.0f
#define KNMS    4096
#define T1      1024
#define T2      1024

typedef __attribute__((ext_vector_type(16))) _Float16     v16h;
typedef __attribute__((ext_vector_type(8)))  float        v8f;
typedef __attribute__((ext_vector_type(4)))  unsigned int v4u;
typedef __attribute__((ext_vector_type(8)))  int          v8i;
typedef __attribute__((ext_vector_type(4)))  int          v4i;

union V16H { v16h v; _Float16 e[16]; };
union V8F  { v8f  v; float    e[8];  };

// WMMA 16-bit A/B slot -> K mapping (lanes<16 hold K{0..7,16..23}, lanes>=16 hold K{8..15,24..31})
__device__ __forceinline__ int kdim_of(int e, int lane) {
  const int v = e >> 1, hf = e & 1;
  return ((lane < 16) ? 0 : 8) + ((v < 4) ? 0 : 16) + ((v & 3) << 1) + hf;
}

// ---------------------------------------------------------------------------
// Kernel 1: per-image filter + order-preserving compaction (dual packed scan).
// ---------------------------------------------------------------------------
__global__ __launch_bounds__(T1)
void yolo_filter_kernel(const float* __restrict__ pred,
                        int*   __restrict__ cnt2,
                        int*   __restrict__ list1,
                        int*   __restrict__ cand_row,
                        float* __restrict__ cand_conf,
                        int*   __restrict__ cand_j) {
  const int b   = blockIdx.x;
  const int tid = threadIdx.x;
  const float* p = pred + (size_t)b * NPRED * ROWSZ;
  int*   L1 = list1     + (size_t)b * KNMS;
  int*   CR = cand_row  + (size_t)b * KNMS;
  float* CC = cand_conf + (size_t)b * KNMS;
  int*   CJ = cand_j    + (size_t)b * KNMS;

  __shared__ int sA[T1];
  __shared__ int sB[T1];

  int base1 = 0, base2 = 0;
  for (int chunk = 0; chunk < NPRED; chunk += T1) {
    const int r = chunk + tid;
    const bool act = (r < NPRED);
    float conf = 0.f;
    int   jm   = 0;
    bool  v1 = false, v2 = false;
    if (act) {
      const float* row = p + (size_t)r * ROWSZ;
      const float obj = row[4];
      v1 = obj > CONF_T;
      if (v1) {
        __builtin_prefetch(row + 5, 0, 0);   // global_prefetch_b8 of the class row
        float m = row[5];
        jm = 0;
        #pragma unroll 4
        for (int c = 1; c < NCLS; ++c) {
          const float v = row[5 + c];
          if (v > m) { m = v; jm = c; }      // first-max == jnp.argmax
        }
        conf = obj * m;                      // == max_c(cls_c * obj), obj > 0
        v2 = conf > CONF_T;
      }
    }

    const int packed = (v1 ? 1 : 0) | ((v2 ? 1 : 0) << 16);
    sA[tid] = packed;
    __syncthreads();
    int* src = sA;
    int* dst = sB;
    for (int off = 1; off < T1; off <<= 1) {
      int v = src[tid];
      if (tid >= off) v += src[tid - off];
      dst[tid] = v;
      __syncthreads();
      int* t = src; src = dst; dst = t;
    }
    const int incl = src[tid];
    const int tot  = src[T1 - 1];
    const int excl = incl - packed;
    const int q1 = base1 + (excl & 0xFFFF);
    const int q2 = base2 + (excl >> 16);
    if (v1 && q1 < KNMS) L1[q1] = r;
    if (v2 && q2 < KNMS) { CR[q2] = r; CC[q2] = conf; CJ[q2] = jm; }
    base1 += (tot & 0xFFFF);
    base2 += (tot >> 16);
    __syncthreads();
  }
  if (tid == 0) cnt2[b] = (base2 < KNMS) ? base2 : KNMS;
}

// Dynamic LDS layout for kernel 2 (bytes) -- see carve order in kernel
#define SMEM2 (KNMS * 4 /*skey*/ + KNMS * 4 /*sidx*/ + 32 * 16 * 2 /*hsw_a*/ + \
               5 * 32 * 16 * 2 /*hsw_b*/ + KNMS * 16 /*boxes*/ +               \
               KNMS * 4 /*area*/ + 1024 * 4 /*kpick*/ + NCLS * 4 /*sout*/ +    \
               16 /*snk*/ + 32 * ROWSZ * 4 /*fstage*/ + KNMS /*keep*/)

// ---------------------------------------------------------------------------
// Kernel 2: bitonic top-K sort + greedy NMS + TDM row-gather + WMMA reduction.
// ---------------------------------------------------------------------------
__global__ __launch_bounds__(T2)
void yolo_nms_kernel(const float* __restrict__ pred,
                     const int*   __restrict__ cnt2,
                     const int*   __restrict__ list1,
                     const int*   __restrict__ cand_row,
                     const float* __restrict__ cand_conf,
                     const int*   __restrict__ cand_j,
                     float*       __restrict__ out) {
  extern __shared__ unsigned char smem[];
  float*    skey  = (float*)smem;                        // KNMS
  int*      sidx  = (int*)(skey + KNMS);                 // KNMS
  _Float16* hsw_a = (_Float16*)(sidx + KNMS);            // 32*16 (lane-major A frag)
  _Float16* hsw_b = hsw_a + 32 * 16;                     // 5*32*16 (tile,lane-major B frag)
  float*    bx1   = (float*)(hsw_b + 5 * 32 * 16);       // KNMS
  float*    by1   = bx1 + KNMS;
  float*    bx2   = by1 + KNMS;
  float*    by2   = bx2 + KNMS;
  float*    sarea = by2 + KNMS;                          // KNMS
  int*      kpick = (int*)(sarea + KNMS);                // 1024
  float*    sout  = (float*)(kpick + 1024);              // NCLS
  int*      snk   = (int*)(sout + NCLS);                 // 4 ints
  float*    fstage= (float*)(snk + 4);                   // 32*ROWSZ staged rows (f32)
  unsigned char* keep = (unsigned char*)(fstage + 32 * ROWSZ); // KNMS

  const int b   = blockIdx.x;
  const int tid = threadIdx.x;
  const float* p  = pred      + (size_t)b * NPRED * ROWSZ;
  const int*   L1 = list1     + (size_t)b * KNMS;
  const int*   CR = cand_row  + (size_t)b * KNMS;
  const float* CC = cand_conf + (size_t)b * KNMS;
  const int*   CJ = cand_j    + (size_t)b * KNMS;
  const int c2 = cnt2[b];
  const int M  = (c2 < KNMS) ? c2 : KNMS;

  // init sort keys (pad with -inf, unique tie-break indices)
  for (int t = tid; t < KNMS; t += T2) {
    if (t < M) { skey[t] = CC[t]; sidx[t] = t; }
    else       { skey[t] = -FLT_MAX; sidx[t] = KNMS + t; }
  }
  __syncthreads();

  // bitonic sort: descending by conf, ties broken by ascending index
  for (int size = 2; size <= KNMS; size <<= 1) {
    for (int stride = size >> 1; stride > 0; stride >>= 1) {
      for (int t = tid; t < KNMS / 2; t += T2) {
        const int lo = ((t & ~(stride - 1)) << 1) | (t & (stride - 1));
        const int hi = lo + stride;
        const bool dir = ((lo & size) == 0);
        const float ka = skey[lo], kb = skey[hi];
        const int   ia = sidx[lo], ib = sidx[hi];
        const bool aFirst = (ka > kb) || (ka == kb && ia < ib);
        if (aFirst != dir) {
          skey[lo] = kb; skey[hi] = ka;
          sidx[lo] = ib; sidx[hi] = ia;
        }
      }
      __syncthreads();
    }
  }

  // build class-offset xyxy boxes for the sorted candidates
  for (int k = tid; k < M; k += T2) {
    const int slot = sidx[k];
    const int r    = CR[slot];
    const float* row = p + (size_t)r * ROWSZ;
    const float x = row[0], y = row[1], w = row[2], h = row[3];
    const float off = (float)CJ[slot] * MAX_WH;
    const float hx = w * 0.5f, hy = h * 0.5f;
    const float X1 = x - hx + off, Y1 = y - hy + off;
    const float X2 = x + hx + off, Y2 = y + hy + off;
    bx1[k] = X1; by1[k] = Y1; bx2[k] = X2; by2[k] = Y2;
    sarea[k] = (X2 - X1) * (Y2 - Y1);
    keep[k] = 1;
  }
  __syncthreads();

  // greedy NMS: pivot i sequential, suppression sweep parallel
  for (int i = 0; i < M; ++i) {
    if (keep[i]) {
      const float X1 = bx1[i], Y1 = by1[i], X2 = bx2[i], Y2 = by2[i];
      const float Ai = sarea[i];
      for (int j = i + 1 + tid; j < M; j += T2) {
        if (!keep[j]) continue;
        const float iw = fminf(bx2[j], X2) - fmaxf(bx1[j], X1);
        const float ih = fminf(by2[j], Y2) - fmaxf(by1[j], Y1);
        const float inter = fmaxf(iw, 0.f) * fmaxf(ih, 0.f);
        const float iou = inter / (sarea[j] + Ai - inter);
        if (iou > IOU_T) keep[j] = 0;
      }
    }
    __syncthreads();
  }

  // cap at MAXDET (reference cumsum trick) and resolve pick = idx1[order]
  if (tid == 0) {
    int nk = 0;
    for (int k = 0; k < M; ++k) {
      if (keep[k]) {
        if (nk < MAXDET) { kpick[nk] = L1[sidx[k]]; ++nk; }
        else             { keep[k] = 0; }
      }
    }
    *snk = nk;
  }
  __syncthreads();
  const int nk = *snk;

  // Accumulate out[c] = sum_k obj(pick_k) * cls(pick_k, c) via WMMA.
  V8F acc[5];
  #pragma unroll
  for (int t = 0; t < 5; ++t)
    #pragma unroll
    for (int e = 0; e < 8; ++e) acc[t].e[e] = 0.f;

  const int nchunk = (nk + 31) >> 5;
  for (int ch = 0; ch < nchunk; ++ch) {
    const int kbase = ch << 5;

    // ---- Stage 32 picked rows (85 f32 each) into LDS ----------------------
#if defined(__gfx1250__) && __has_builtin(__builtin_amdgcn_tensor_load_to_lds)
    if (tid < 32) {  // wave 0 issues the DMA (tensor ops issue once per wave)
      #pragma unroll
      for (int half = 0; half < 2; ++half) {
        int rows = nk - kbase - half * 16;
        rows = rows < 0 ? 0 : (rows > 16 ? 16 : rows);
        if (rows > 0) {
          // D# group 0: count=1, gather_mode=1 (16-bit indices), lds/global addr
          const unsigned lds_off =
              (unsigned)(size_t)(const void*)(fstage + half * 16 * ROWSZ);
          const unsigned long long ga = (unsigned long long)(const void*)p;
          v4u g0;
          g0.x = (unsigned)__builtin_amdgcn_readfirstlane((int)(1u | (1u << 31)));
          g0.y = (unsigned)__builtin_amdgcn_readfirstlane((int)lds_off);
          g0.z = (unsigned)__builtin_amdgcn_readfirstlane((int)(unsigned)(ga & 0xFFFFFFFFu));
          g0.w = (unsigned)__builtin_amdgcn_readfirstlane(
                     (int)(((unsigned)((ga >> 32) & 0x01FFFFFFu)) | (2u << 30)));
          // D# group 1: data_size=4B, tensor_dim0=85, tensor_dim1=NPRED,
          //             tile_dim0=85, tile_dim1=rows, dim0 stride=85
          v8i g1;
          g1[0] = __builtin_amdgcn_readfirstlane((int)(2u << 16));
          g1[1] = __builtin_amdgcn_readfirstlane((int)(((unsigned)ROWSZ & 0xFFFFu) << 16));
          g1[2] = __builtin_amdgcn_readfirstlane((int)(((unsigned)NPRED & 0xFFFFu) << 16));
          g1[3] = __builtin_amdgcn_readfirstlane((int)(((unsigned)ROWSZ << 16) |
                                                       ((unsigned)NPRED >> 16)));
          g1[4] = __builtin_amdgcn_readfirstlane(rows);
          g1[5] = __builtin_amdgcn_readfirstlane((int)ROWSZ);
          g1[6] = 0;
          g1[7] = 0;
          // D# groups 2/3: 16 x 16-bit gathered row indices
          unsigned idx16[16];
          #pragma unroll
          for (int i = 0; i < 16; ++i)
            idx16[i] = (i < rows) ? (unsigned)kpick[kbase + half * 16 + i] : 0u;
          v4i g2, g3;
          #pragma unroll
          for (int i = 0; i < 4; ++i) {
            g2[i] = __builtin_amdgcn_readfirstlane(
                        (int)(idx16[2 * i] | (idx16[2 * i + 1] << 16)));
            g3[i] = __builtin_amdgcn_readfirstlane(
                        (int)(idx16[8 + 2 * i] | (idx16[9 + 2 * i] << 16)));
          }
          // this toolchain exposes the 6-arg form: 4 D# groups + extra
          // 8-dword group (unused here, zero) + cache-policy immediate
          const v8i g4 = {0, 0, 0, 0, 0, 0, 0, 0};
          __builtin_amdgcn_tensor_load_to_lds(g0, g1, g2, g3, g4, 0);
        }
      }
      __builtin_amdgcn_s_wait_tensorcnt(0);
    }
#else
    for (int t = tid; t < 32 * ROWSZ; t += T2) {
      const int kk = t / ROWSZ;
      if (kbase + kk < nk)
        fstage[t] = p[(size_t)kpick[kbase + kk] * ROWSZ + (t % ROWSZ)];
    }
#endif
    __syncthreads();

    // ---- Convert f32 -> f16 directly into WMMA lane-major fragment layout -
    for (int t = tid; t < 32 * 16; t += T2) {          // A: obj broadcast
      const int lane = t >> 4, e = t & 15;
      const int kd = kdim_of(e, lane);
      hsw_a[t] = (kbase + kd < nk) ? (_Float16)fstage[kd * ROWSZ + 4]
                                   : (_Float16)0.f;
    }
    for (int t = tid; t < 5 * 32 * 16; t += T2) {      // B: class tiles
      const int e = t & 15, lane = (t >> 4) & 31, tt = t >> 9;
      const int kd = kdim_of(e, lane);
      const int c = tt * 16 + (lane & 15);
      hsw_b[t] = (kbase + kd < nk) ? (_Float16)fstage[kd * ROWSZ + 5 + c]
                                   : (_Float16)0.f;
    }
    __syncthreads();

    // ---- Matrix op: wave 0, full EXEC; frags are two ds_load_b128 each ----
    if (tid < 32) {
      const v16h a = *(const v16h*)(hsw_a + tid * 16);
      #pragma unroll
      for (int t = 0; t < 5; ++t) {
        const v16h bf = *(const v16h*)(hsw_b + (t * 32 + tid) * 16);
        acc[t].v = __builtin_amdgcn_wmma_f32_16x16x32_f16(
            false, a, false, bf, (short)0, acc[t].v, false, false);
      }
    }
    __syncthreads();
  }

  // C layout: VGPR0, lanes 0..15 hold (M=0, N=lane) -> row 0 of each tile
  if (tid < 16) {
    #pragma unroll
    for (int t = 0; t < 5; ++t) sout[t * 16 + tid] = acc[t].e[0];
  }
  __syncthreads();
  if (tid < NCLS) out[b * NCLS + tid] = sout[tid];
}

// ---------------------------------------------------------------------------
extern "C" void kernel_launch(void* const* d_in, const int* in_sizes, int n_in,
                              void* d_out, int out_size, void* d_ws, size_t ws_size,
                              hipStream_t stream) {
  (void)in_sizes; (void)n_in; (void)out_size; (void)ws_size;
  const float* pred = (const float*)d_in[0];
  float* out = (float*)d_out;

  // workspace carve-out (~2.1 MB)
  int*   cnt2      = (int*)d_ws;
  int*   list1     = cnt2 + BATCH;
  int*   cand_row  = list1 + (size_t)BATCH * KNMS;
  float* cand_conf = (float*)(cand_row + (size_t)BATCH * KNMS);
  int*   cand_j    = (int*)(cand_conf + (size_t)BATCH * KNMS);

  // ~137 KB dynamic LDS per workgroup (CDNA5 WGP has 320 KB)
  (void)hipFuncSetAttribute((const void*)yolo_nms_kernel,
                            hipFuncAttributeMaxDynamicSharedMemorySize, (int)SMEM2);

  hipLaunchKernelGGL(yolo_filter_kernel, dim3(BATCH), dim3(T1), 0, stream,
                     pred, cnt2, list1, cand_row, cand_conf, cand_j);
  hipLaunchKernelGGL(yolo_nms_kernel, dim3(BATCH), dim3(T2), SMEM2, stream,
                     pred, cnt2, list1, cand_row, cand_conf, cand_j, out);
}
